// Vandermonde_68667937128993
// MI455X (gfx1250) — matrix operand order repaired and checked by reading, not compiled, not run on hardware
//
#include <hip/hip_runtime.h>

// CDNA5 / gfx1250 WMMA types
typedef __attribute__((ext_vector_type(16))) _Float16 v16h;
typedef __attribute__((ext_vector_type(8)))  float    v8f;

#define BATCH   2
#define SEQ     512
#define NHEADS  8
#define QKD     16
#define HEADDIM 128
#define DINNER  (NHEADS * HEADDIM)      // 1024
#define EPSF    1e-3f
#define STDF    0.0078125f              // 1/sqrt(2*512*16) = 1/128 exactly

// Grid: B * NHEADS * (SEQ/16) = 512 blocks, one 16-row t-tile per block.
// Block: 256 threads = 8 wave32; wave w handles h-columns [16w, 16w+16).
__global__ __launch_bounds__(256) void vander_wmma_kernel(
    const float* __restrict__ v,
    const float* __restrict__ q,
    const float* __restrict__ k,
    float* __restrict__ out)
{
    // 16x32 S-tile in f16, pre-arranged in the WMMA A-fragment layout
    // (contiguous k order, same as B; not the ISA table order): lane = 16*g + M, half j -> K as below.
    __shared__ __align__(32) _Float16 sfrag[32 * 16];

    const int tid  = threadIdx.x;
    const int lane = tid & 31;
    const int wave = tid >> 5;          // 0..7
    const int hw   = wave << 4;         // h-chunk base

    const int bidx = blockIdx.x;
    const int t0   = (bidx & 31) << 4;  // t-tile base
    const int n    = (bidx >> 5) & 7;   // head
    const int b    = bidx >> 8;         // batch

    // S-element assignment: this thread computes S[t, lA] and S[t, lA+16]
    const int t  = tid & 15;
    const int lA = tid >> 4;            // 0..15
    const float tpos = (float)(t0 + t);

    // Cache q angles for row t, pre-scaled by EPS. Note: the hardware
    // v_cos_f32 argument is in turns (cos(2*pi*x)), so the reference's
    // 2*pi*EPS factor reduces to just EPS here.
    float eq[QKD];
    {
        const float* qp = q + ((size_t)(b * SEQ + t0 + t) * NHEADS + n) * QKD;
        #pragma unroll
        for (int d = 0; d < QKD; ++d) eq[d] = EPSF * qp[d];
    }

    v8f acc = {};                       // f32 C/D accumulator (8 VGPRs)

    // B-fragment (contiguous k order, matching A): lane -> N = lane%16, K group = 16*(lane/16)
    const int nB  = lane & 15;
    const int kgB = (lane >> 4) << 4;
    const float* vbase = v + (size_t)(b * SEQ) * DINNER + n * HEADDIM + hw + nB;

    for (int l0 = 0; l0 < SEQ; l0 += 32) {
        __syncthreads();                // previous iteration's A reads done

        // Cooperatively build the 16x32 S tile (2 elements per thread),
        // 32 v_cos_f32 per element — this is the arithmetic roofline term.
        #pragma unroll
        for (int i = 0; i < 2; ++i) {
            const int l = lA + 16 * i;
            const float lpos = (float)(l0 + l);
            const float* kp = k + ((size_t)(b * SEQ + l0 + l) * NHEADS + n) * QKD;
            float s = 0.f;
            #pragma unroll
            for (int d = 0; d < QKD; ++d) {
                s += __builtin_amdgcn_cosf(eq[d] * lpos);         // q term
                s -= __builtin_amdgcn_cosf(EPSF * kp[d] * tpos);  // k term
            }
            // A-fragment address: l in [0,16)->g=0,j=l ; [16,32)->g=1,j=l-16
            const int g = l >> 4;
            const int j = l & 15;
            sfrag[(16 * g + t) * 16 + j] = (_Float16)s;
        }

        __syncthreads();                // S tile visible to all 8 waves

        // A fragment: 16 consecutive halves per lane -> 2x ds_load_b128
        v16h afrag = *(const v16h*)(&sfrag[lane * 16]);

        // B fragment: V[l0+K][n*128 + hw + N] (fp32 -> f16), K = kgB + j
        v16h bfrag;
        const float* vp = vbase + (size_t)(l0 + kgB) * DINNER;
        #pragma unroll
        for (int j = 0; j < 16; ++j)
            bfrag[j] = (_Float16)vp[(size_t)j * DINNER];

        // D = A(16x32,f16) x B(32x16,f16) + C(f32)
        acc = __builtin_amdgcn_wmma_f32_16x16x32_f16(
            false, afrag, false, bfrag, (short)0, acc, false, false);
    }

    // Epilogue: C/D layout -> lane gives N, VGPR r gives M = r + 8*(lane/16).
    // out = (1/128) * acc + residual(v)
    #pragma unroll
    for (int r = 0; r < 8; ++r) {
        const int m = r + ((lane >> 4) << 3);
        const size_t idx = (size_t)(b * SEQ + t0 + m) * DINNER
                         + n * HEADDIM + hw + nB;
        out[idx] = STDF * acc[r] + v[idx];
    }
}

extern "C" void kernel_launch(void* const* d_in, const int* in_sizes, int n_in,
                              void* d_out, int out_size, void* d_ws, size_t ws_size,
                              hipStream_t stream) {
    const float* v = (const float*)d_in[0];
    const float* q = (const float*)d_in[1];
    const float* k = (const float*)d_in[2];
    float* out = (float*)d_out;

    dim3 grid(BATCH * NHEADS * (SEQ / 16));   // 512 blocks
    dim3 block(256);                          // 8 wave32
    vander_wmma_kernel<<<grid, block, 0, stream>>>(v, q, k, out);
}
